// PointRend_38972533244638
// MI455X (gfx1250) — matrix operand order repaired and checked by reading, not compiled, not run on hardware
//
#include <hip/hip_runtime.h>
#include <hip/hip_bf16.h>

// Problem constants (from reference setup_inputs)
#define B_    8
#define NC    19
#define HC    32
#define WC    64
#define CF    256
#define HF    128
#define WF    256
#define HW    (HF*WF)          // 32768
#define HID   256
#define NPTS  2048
#define KIN   275              // CF + NC
#define KPAD  288              // 9 k-tiles of 32
#define XSTR  304              // LDS column stride (bf16 elems), 608B: 32B aligned, conflict-padded
#define HSTR  272              // 544B stride for hidden activations

typedef __attribute__((ext_vector_type(16))) __bf16 v16bf;
typedef __attribute__((ext_vector_type(8)))  float  v8f;

__device__ __forceinline__ unsigned short f2bf(float f) {
  unsigned u = __float_as_uint(f);
  u += 0x7FFFu + ((u >> 16) & 1u);   // round-to-nearest-even
  return (unsigned short)(u >> 16);
}

// ---------------------------------------------------------------------------
// Kernel 1: bilinear upsample (align_corners=True) + softmax-uncertainty keys
// block = one fine row: threadIdx.x = x (coalesced stores per channel)
// ---------------------------------------------------------------------------
__global__ __launch_bounds__(WF) void k_upsample(const float* __restrict__ coarse,
                                                 float* __restrict__ out,
                                                 unsigned* __restrict__ keys) {
  const int b = blockIdx.x / HF;
  const int y = blockIdx.x % HF;
  const int x = threadIdx.x;
  float ysrc = (float)y * (float)(HC - 1) / (float)(HF - 1);
  float xsrc = (float)x * (float)(WC - 1) / (float)(WF - 1);
  int y0 = (int)floorf(ysrc); y0 = min(max(y0, 0), HC - 1);
  int x0 = (int)floorf(xsrc); x0 = min(max(x0, 0), WC - 1);
  int y1 = min(y0 + 1, HC - 1), x1 = min(x0 + 1, WC - 1);
  float wy = ysrc - (float)y0, wx = xsrc - (float)x0;

  float vals[NC];
  float vmax = -1e30f;
#pragma unroll
  for (int c = 0; c < NC; ++c) {
    const float* p = coarse + ((size_t)(b * NC + c)) * (HC * WC);
    float v00 = p[y0 * WC + x0], v01 = p[y0 * WC + x1];
    float v10 = p[y1 * WC + x0], v11 = p[y1 * WC + x1];
    float r0 = v00 * (1.f - wy) + v10 * wy;   // rows step (matches reference order)
    float r1 = v01 * (1.f - wy) + v11 * wy;
    float v  = r0 * (1.f - wx) + r1 * wx;
    vals[c] = v;
    vmax = fmaxf(vmax, v);
    out[((size_t)(b * NC + c)) * HW + y * WF + x] = v;
  }
  float se = 0.f;
#pragma unroll
  for (int c = 0; c < NC; ++c) se += expf(vals[c] - vmax);
  float unc = -1.0f / se;                    // == -max(softmax)
  unsigned u = __float_as_uint(unc);
  unsigned key = (u >> 31) ? ~u : (u | 0x80000000u);   // monotone float->u32 map
  keys[(size_t)b * HW + y * WF + x] = key;
}

// ---------------------------------------------------------------------------
// Kernel 2: per-batch radix select of the 2048th-largest key (exact threshold)
// ---------------------------------------------------------------------------
__global__ __launch_bounds__(256) void k_select(const unsigned* __restrict__ keys,
                                                unsigned* __restrict__ thr,
                                                unsigned* __restrict__ cgt,
                                                unsigned* __restrict__ ctrg,
                                                unsigned* __restrict__ ctre) {
  __shared__ unsigned hist[256];
  __shared__ unsigned s_prefix, s_k, s_cgt;
  const int b = blockIdx.x, tid = threadIdx.x;
  if (tid == 0) { s_prefix = 0u; s_k = NPTS; s_cgt = 0u; ctrg[b] = 0u; ctre[b] = 0u; }
  const unsigned* kb = keys + (size_t)b * HW;
  for (int shift = 24; shift >= 0; shift -= 8) {
    hist[tid] = 0u;
    __syncthreads();
    unsigned mask_hi = (shift == 24) ? 0u : (0xFFFFFFFFu << (shift + 8));
    unsigned prefix  = s_prefix;
    for (int i = tid; i < HW; i += 256) {
      unsigned k = kb[i];
      if ((k & mask_hi) == prefix) atomicAdd(&hist[(k >> shift) & 255u], 1u);
    }
    __syncthreads();
    if (tid == 0) {
      unsigned cum = 0, k = s_k;
      for (int d = 255; d >= 0; --d) {
        if (cum + hist[d] >= k) {
          s_prefix = prefix | ((unsigned)d << shift);
          s_cgt += cum;
          s_k = k - cum;
          break;
        }
        cum += hist[d];
      }
    }
    __syncthreads();
  }
  if (tid == 0) { thr[b] = s_prefix; cgt[b] = s_cgt; }
}

// ---------------------------------------------------------------------------
// Kernel 3: compact indices of top-NPTS keys into a per-batch list
// ---------------------------------------------------------------------------
__global__ __launch_bounds__(256) void k_compact(const unsigned* __restrict__ keys,
                                                 const unsigned* __restrict__ thr,
                                                 const unsigned* __restrict__ cgt,
                                                 unsigned* __restrict__ ctrg,
                                                 unsigned* __restrict__ ctre,
                                                 int* __restrict__ list) {
  const int gid = blockIdx.x * 256 + threadIdx.x;   // grid covers B_*HW exactly
  const int b = gid / HW, i = gid % HW;
  unsigned k = keys[gid], t = thr[b];
  if (k > t) {
    unsigned pos = atomicAdd(&ctrg[b], 1u);
    list[b * NPTS + pos] = i;
  } else if (k == t) {
    unsigned pos = cgt[b] + atomicAdd(&ctre[b], 1u);
    if (pos < NPTS) list[b * NPTS + pos] = i;
  }
}

// ---------------------------------------------------------------------------
// Kernel 4: repack fp32 weights to bf16 WMMA A-fragment order.
// A 16x32 k order chosen here (contiguous; B reads use the same): lane l (m = l&15, h = l>>4), element p in 0..15:
//   K = p + 16*h
// Fragment storage: frag[tile][lane][16], tile = mt*ktiles + kt  (zero padded).
// ---------------------------------------------------------------------------
__global__ __launch_bounds__(256) void k_repack(const float* __restrict__ w1,
                                                const float* __restrict__ w2,
                                                const float* __restrict__ w3,
                                                unsigned short* __restrict__ w1f,
                                                unsigned short* __restrict__ w2f,
                                                unsigned short* __restrict__ w3f) {
  const int n1 = 16 * 9 * 512, n2 = 16 * 8 * 512, n3 = 2 * 8 * 512;
  int id = blockIdx.x * 256 + threadIdx.x;
  const float* W; unsigned short* F; int Mreal, Kreal, ktiles;
  if (id < n1)            { W = w1; F = w1f; Mreal = HID; Kreal = KIN; ktiles = 9; }
  else if (id < n1 + n2)  { id -= n1; W = w2; F = w2f; Mreal = HID; Kreal = HID; ktiles = 8; }
  else if (id < n1+n2+n3) { id -= n1 + n2; W = w3; F = w3f; Mreal = NC; Kreal = HID; ktiles = 8; }
  else return;
  int e = id & 511, tile = id >> 9;
  int kt = tile % ktiles, mt = tile / ktiles;
  int ln = e >> 4, p = e & 15;
  int m  = mt * 16 + (ln & 15);
  int h  = ln >> 4;
  int kk = kt * 32 + p + 16 * h;
  float v = (m < Mreal && kk < Kreal) ? W[m * Kreal + kk] : 0.f;
  F[(size_t)tile * 512 + e] = f2bf(v);
}

// ---------------------------------------------------------------------------
// Kernel 5: gather point features + 3-layer MLP via v_wmma_f32_16x16x32_bf16.
// Block = 32 points (2 N-tiles), 8 waves. Each wave owns 2 M-tiles x BOTH
// N-tiles, so every A-fragment (weight) load feeds two WMMAs -> halves the
// L2 weight-fragment traffic vs a 1-N-tile-per-wave mapping.
// B-fragment (bf16 32x16): lane l -> col = l&15, K base = 16*(l>>4): one
// contiguous 32B v16bf read from column-major LDS.
// ---------------------------------------------------------------------------
__global__ __launch_bounds__(256) void k_mlp(const float* __restrict__ fine,
                                             const float* __restrict__ cup,
                                             const int* __restrict__ list,
                                             const unsigned short* __restrict__ w1f,
                                             const unsigned short* __restrict__ w2f,
                                             const unsigned short* __restrict__ w3f,
                                             const float* __restrict__ b1,
                                             const float* __restrict__ b2,
                                             const float* __restrict__ b3,
                                             float* __restrict__ logits) {
  __shared__ __align__(32) unsigned short X[32 * XSTR];    // layer-1 input (also reused as H2)
  __shared__ __align__(32) unsigned short H1[32 * HSTR];
  __shared__ int   px0[32], px1[32], py0[32], py1[32];
  __shared__ float pwx[32], pwy[32];
  unsigned short* H2 = X;                                  // X dead after layer 1

  const int tid = threadIdx.x, blk = blockIdx.x;
  const int b = (blk * 32) / NPTS;                         // 32 | NPTS -> one batch per block

  if (tid < 32) {
    int idx = list[blk * 32 + tid];
    int yi = idx / WF, xi = idx - yi * WF;
    // replicate reference fp arithmetic for coords -> sample positions
    float xn = (float)xi / (float)(WF - 1) * 2.f - 1.f;
    float yn = (float)yi / (float)(HF - 1) * 2.f - 1.f;
    float ix = (xn + 1.f) * 0.5f * (float)(WF - 1);
    float iy = (yn + 1.f) * 0.5f * (float)(HF - 1);
    int x0 = min(max((int)floorf(ix), 0), WF - 1);
    int y0 = min(max((int)floorf(iy), 0), HF - 1);
    px0[tid] = x0; px1[tid] = min(x0 + 1, WF - 1);
    py0[tid] = y0; py1[tid] = min(y0 + 1, HF - 1);
    pwx[tid] = ix - (float)x0; pwy[tid] = iy - (float)y0;
  }
  __syncthreads();

  // Gather 275 channels (fine 0..255, coarse 256..274, pad to 288) -> LDS bf16
  for (int j = tid; j < 32 * KPAD; j += 256) {
    int p = j / KPAD, k = j - p * KPAD;
    float v = 0.f;
    if (k < KIN) {
      const float* base = (k < CF) ? (fine + ((size_t)b * CF + k) * HW)
                                   : (cup  + ((size_t)b * NC + (k - CF)) * HW);
      int x0 = px0[p], x1 = px1[p], y0 = py0[p], y1 = py1[p];
      float wx = pwx[p], wy = pwy[p];
      float v00 = base[y0 * WF + x0], v01 = base[y0 * WF + x1];
      float v10 = base[y1 * WF + x0], v11 = base[y1 * WF + x1];
      v = v00 * (1.f - wx) * (1.f - wy) + v01 * wx * (1.f - wy)
        + v10 * (1.f - wx) * wy        + v11 * wx * wy;
    }
    X[p * XSTR + k] = f2bf(v);
  }
  __syncthreads();

  const int lane = tid & 31, wave = tid >> 5;
  const int col = lane & 15, half = lane >> 4;

  // ---- Layer 1: [256 x 288] x [288 x 32]; wave owns mtiles {2w, 2w+1} ----
  v8f a00 = {}, a01 = {}, a10 = {}, a11 = {};              // a[mt][ntile]
  for (int kt = 0; kt < 9; ++kt) {
    v16bf bf0 = *(const v16bf*)&X[col * XSTR + kt * 32 + half * 16];
    v16bf bf1 = *(const v16bf*)&X[(16 + col) * XSTR + kt * 32 + half * 16];
    const unsigned short* wb = w1f + ((size_t)(wave * 2) * 9 + kt) * 512 + lane * 16;
    v16bf f0 = *(const v16bf*)(wb);
    v16bf f1 = *(const v16bf*)(wb + 9 * 512);
    a00 = __builtin_amdgcn_wmma_f32_16x16x32_bf16(false, f0, false, bf0, (short)0, a00, false, false);
    a01 = __builtin_amdgcn_wmma_f32_16x16x32_bf16(false, f0, false, bf1, (short)0, a01, false, false);
    a10 = __builtin_amdgcn_wmma_f32_16x16x32_bf16(false, f1, false, bf0, (short)0, a10, false, false);
    a11 = __builtin_amdgcn_wmma_f32_16x16x32_bf16(false, f1, false, bf1, (short)0, a11, false, false);
  }
#pragma unroll
  for (int i = 0; i < 4; ++i) {
    v8f a = (i == 0) ? a00 : (i == 1) ? a01 : (i == 2) ? a10 : a11;
    int mt = wave * 2 + (i >> 1);
    int pt = (i & 1) * 16 + col;
#pragma unroll
    for (int r = 0; r < 8; ++r) {
      int c = mt * 16 + r + 8 * half;                       // D layout: m = r + 8*(lane/16)
      H1[pt * HSTR + c] = f2bf(fmaxf(a[r] + b1[c], 0.f));
    }
  }
  __syncthreads();

  // ---- Layer 2: [256 x 256] x [256 x 32] ----
  a00 = (v8f){}; a01 = (v8f){}; a10 = (v8f){}; a11 = (v8f){};
  for (int kt = 0; kt < 8; ++kt) {
    v16bf bf0 = *(const v16bf*)&H1[col * HSTR + kt * 32 + half * 16];
    v16bf bf1 = *(const v16bf*)&H1[(16 + col) * HSTR + kt * 32 + half * 16];
    const unsigned short* wb = w2f + ((size_t)(wave * 2) * 8 + kt) * 512 + lane * 16;
    v16bf f0 = *(const v16bf*)(wb);
    v16bf f1 = *(const v16bf*)(wb + 8 * 512);
    a00 = __builtin_amdgcn_wmma_f32_16x16x32_bf16(false, f0, false, bf0, (short)0, a00, false, false);
    a01 = __builtin_amdgcn_wmma_f32_16x16x32_bf16(false, f0, false, bf1, (short)0, a01, false, false);
    a10 = __builtin_amdgcn_wmma_f32_16x16x32_bf16(false, f1, false, bf0, (short)0, a10, false, false);
    a11 = __builtin_amdgcn_wmma_f32_16x16x32_bf16(false, f1, false, bf1, (short)0, a11, false, false);
  }
#pragma unroll
  for (int i = 0; i < 4; ++i) {
    v8f a = (i == 0) ? a00 : (i == 1) ? a01 : (i == 2) ? a10 : a11;
    int mt = wave * 2 + (i >> 1);
    int pt = (i & 1) * 16 + col;
#pragma unroll
    for (int r = 0; r < 8; ++r) {
      int c = mt * 16 + r + 8 * half;
      H2[pt * XSTR + c] = f2bf(fmaxf(a[r] + b2[c], 0.f));   // H2 aliases X (X is dead)
    }
  }
  __syncthreads();

  // ---- Layer 3: [32(pad of 19) x 256] x [256 x 32]; 4 tiles on waves 0..3 ----
  if (wave < 4) {
    const int mt = wave & 1, nt = wave >> 1;
    const int pt = nt * 16 + col;
    v8f acc = {};
    for (int kt = 0; kt < 8; ++kt) {
      v16bf bf = *(const v16bf*)&H2[pt * XSTR + kt * 32 + half * 16];
      v16bf af = *(const v16bf*)&w3f[((size_t)mt * 8 + kt) * 512 + lane * 16];
      acc = __builtin_amdgcn_wmma_f32_16x16x32_bf16(false, af, false, bf, (short)0, acc, false, false);
    }
#pragma unroll
    for (int r = 0; r < 8; ++r) {
      int c = mt * 16 + r + 8 * half;
      if (c < NC) {
        int li = blk * 32 + pt - b * NPTS;                  // point index within batch
        logits[((size_t)b * NC + c) * NPTS + li] = acc[r] + b3[c];
      }
    }
  }
}

// ---------------------------------------------------------------------------
// Kernel 6: scatter refined point logits into the upsampled output
// ---------------------------------------------------------------------------
__global__ __launch_bounds__(256) void k_scatter(const int* __restrict__ list,
                                                 const float* __restrict__ logits,
                                                 float* __restrict__ out) {
  int gid = blockIdx.x * 256 + threadIdx.x;
  if (gid >= B_ * NC * NPTS) return;
  int i = gid % NPTS;
  int c = (gid / NPTS) % NC;
  int b = gid / (NPTS * NC);
  int idx = list[b * NPTS + i];
  out[((size_t)b * NC + c) * HW + idx] = logits[gid];       // logits flat == [b][c][i]
}

// ---------------------------------------------------------------------------
// Workspace layout (bytes, 256-aligned); total ~2.66 MB
// ---------------------------------------------------------------------------
static const size_t OFF_KEYS   = 0;                                  // B*HW u32   = 1 MiB
static const size_t OFF_THR    = OFF_KEYS   + (size_t)B_ * HW * 4;
static const size_t OFF_CGT    = OFF_THR    + 256;
static const size_t OFF_CTRG   = OFF_CGT    + 256;
static const size_t OFF_CTRE   = OFF_CTRG   + 256;
static const size_t OFF_LIST   = OFF_CTRE   + 256;                   // B*NPTS i32
static const size_t OFF_LOGITS = OFF_LIST   + (size_t)B_ * NPTS * 4; // B*NC*NPTS f32
static const size_t OFF_W1F    = OFF_LOGITS + (size_t)B_ * NC * NPTS * 4;
static const size_t OFF_W2F    = OFF_W1F    + (size_t)16 * 9 * 512 * 2;
static const size_t OFF_W3F    = OFF_W2F    + (size_t)16 * 8 * 512 * 2;

extern "C" void kernel_launch(void* const* d_in, const int* in_sizes, int n_in,
                              void* d_out, int out_size, void* d_ws, size_t ws_size,
                              hipStream_t stream) {
  const float* coarse = (const float*)d_in[0];
  const float* fine   = (const float*)d_in[1];
  const float* w1 = (const float*)d_in[2];
  const float* b1 = (const float*)d_in[3];
  const float* w2 = (const float*)d_in[4];
  const float* b2 = (const float*)d_in[5];
  const float* w3 = (const float*)d_in[6];
  const float* b3 = (const float*)d_in[7];
  float* out = (float*)d_out;

  char* ws = (char*)d_ws;
  unsigned* keys = (unsigned*)(ws + OFF_KEYS);
  unsigned* thr  = (unsigned*)(ws + OFF_THR);
  unsigned* cgt  = (unsigned*)(ws + OFF_CGT);
  unsigned* ctrg = (unsigned*)(ws + OFF_CTRG);
  unsigned* ctre = (unsigned*)(ws + OFF_CTRE);
  int*      list = (int*)     (ws + OFF_LIST);
  float*    lgts = (float*)   (ws + OFF_LOGITS);
  unsigned short* w1f = (unsigned short*)(ws + OFF_W1F);
  unsigned short* w2f = (unsigned short*)(ws + OFF_W2F);
  unsigned short* w3f = (unsigned short*)(ws + OFF_W3F);

  k_upsample<<<B_ * HF, WF, 0, stream>>>(coarse, out, keys);
  k_repack  <<<(16*9*512 + 16*8*512 + 2*8*512) / 256, 256, 0, stream>>>(w1, w2, w3, w1f, w2f, w3f);
  k_select  <<<B_, 256, 0, stream>>>(keys, thr, cgt, ctrg, ctre);
  k_compact <<<B_ * HW / 256, 256, 0, stream>>>(keys, thr, cgt, ctrg, ctre, list);
  k_mlp     <<<B_ * NPTS / 32, 256, 0, stream>>>(fine, out, list, w1f, w2f, w3f, b1, b2, b3, lgts);
  k_scatter <<<(B_ * NC * NPTS + 255) / 256, 256, 0, stream>>>(list, lgts, out);
}
